// MambaTimeEncoder_26903675142270
// MI455X (gfx1250) — matrix operand order repaired and checked by reading, not compiled, not run on hardware
//
#include <hip/hip_runtime.h>
#include <hip/hip_bf16.h>
#include <math.h>

// ---------------------------------------------------------------------------
// MambaTimeEncoder forward for MI455X (gfx1250, wave32, WMMA f16 16x16x32)
// ---------------------------------------------------------------------------

namespace {
constexpr int B_   = 8;
constexpr int L_   = 2048;
constexpr int FIN_ = 16;
constexpr int DM_  = 256;   // d_model
constexpr int DI_  = 512;   // d_inner
constexpr int DS_  = 16;    // d_state
constexpr int DTR_ = 16;    // dt_rank
constexpr int XDB_ = DTR_ + 2 * DS_;  // 48
constexpr int M_   = B_ * L_;         // 16384 tokens
}

typedef __attribute__((ext_vector_type(16))) _Float16 v16h;
typedef __attribute__((ext_vector_type(8)))  float    v8f;

__device__ __forceinline__ float siluf(float x)     { return x / (1.0f + __expf(-x)); }
__device__ __forceinline__ float softplusf(float x) { return (x > 20.0f) ? x : log1pf(__expf(x)); }

static inline int cdiv(int a, int b) { return (a + b - 1) / b; }

// Unconditional 8-float -> 8-half fragment fill (two global_load_b128 + cvt_pk)
__device__ __forceinline__ void ld8(v16h& f, int base, const float* p) {
  const float4 x = *(const float4*)p;
  const float4 y = *(const float4*)(p + 4);
  f[base + 0] = (_Float16)x.x; f[base + 1] = (_Float16)x.y;
  f[base + 2] = (_Float16)x.z; f[base + 3] = (_Float16)x.w;
  f[base + 4] = (_Float16)y.x; f[base + 5] = (_Float16)y.y;
  f[base + 6] = (_Float16)y.z; f[base + 7] = (_Float16)y.w;
}

// Branchless guarded fill: loads from a valid (clamped) address, zero-selects.
__device__ __forceinline__ void ld8g(v16h& f, int base, const float* p, bool ok) {
  const float4 x = *(const float4*)p;
  const float4 y = *(const float4*)(p + 4);
  f[base + 0] = (_Float16)(ok ? x.x : 0.0f); f[base + 1] = (_Float16)(ok ? x.y : 0.0f);
  f[base + 2] = (_Float16)(ok ? x.z : 0.0f); f[base + 3] = (_Float16)(ok ? x.w : 0.0f);
  f[base + 4] = (_Float16)(ok ? y.x : 0.0f); f[base + 5] = (_Float16)(ok ? y.y : 0.0f);
  f[base + 6] = (_Float16)(ok ? y.z : 0.0f); f[base + 7] = (_Float16)(ok ? y.w : 0.0f);
}

// ---------------------------------------------------------------------------
// WMMA GEMM: C[M,N] = epilogue(A[M,K] * W[N,K]^T + bias)
//   One wave = 16 x (16*NT) tile, K chunks of 32, A fragment reused NT times.
//   EPI:   0 none, 1 silu, 2 softplus, 3 C = C*mask + v*mask (masked residual)
//   NT:    subtiles (compile-time; grid.y * 16*NT covers N exactly-or-over)
//   GUARD: row bounds needed (only fusion GEMM, M=8)
//   KTAIL: K guards needed (only K=16 GEMMs); else K % 32 == 0 required
// ---------------------------------------------------------------------------
template <int EPI, int NT, bool GUARD, bool KTAIL>
__global__ __launch_bounds__(32) void gemm16_wmma(
    const float* __restrict__ A, int lda,
    const float* __restrict__ W,          // [N,K] row-major
    const float* __restrict__ bias,       // [N] or nullptr
    const int*   __restrict__ maskI,      // [M] (EPI==3) or nullptr
    float* C, int ldc, int M, int N, int K)
{
  const int lane  = threadIdx.x & 31;
  const int lo    = lane & 15;
  const int hi    = lane >> 4;
  const int tileM = blockIdx.x << 4;
  const int n0    = blockIdx.y * (16 * NT);

  v8f acc[NT];
#pragma unroll
  for (int s = 0; s < NT; ++s)
#pragma unroll
    for (int i = 0; i < 8; ++i) acc[s][i] = 0.0f;

  // Row for the A fragment (clamped when GUARD so loads are always valid).
  int arow_i = tileM + lo;
  if (GUARD) arow_i = (arow_i < M) ? arow_i : (M - 1);
  const bool rok    = GUARD ? (tileM + lo < M) : true;
  const float* arow = A + (size_t)arow_i * lda;

  for (int kt = 0; kt < K; kt += 32) {
    // A fragment: lane<16 -> K {kt+0..7, kt+16..23}; lane>=16 -> {kt+8..15, kt+24..31}
    v16h af;
    if (KTAIL || GUARD) {
      const int ka = kt + hi * 8;
      const int kb = kt + 16 + hi * 8;
      const bool oka = rok && (ka < K);
      const bool okb = rok && (kb < K);
      ld8g(af, 0, arow + (oka ? ka : 0), oka);
      ld8g(af, 8, arow + (okb ? kb : 0), okb);
    } else {
      ld8(af, 0, arow + kt + hi * 8);
      ld8(af, 8, arow + kt + 16 + hi * 8);
      if (kt + 32 < K) __builtin_prefetch(arow + kt + 32, 0, 3);
    }

    // B fragments for all subtiles: lane<16 -> K {kt+0..7, kt+16..23}; lane>=16 -> {kt+8..15, kt+24..31}
    v16h bf[NT];
#pragma unroll
    for (int s = 0; s < NT; ++s) {
      const float* wrow = W + (size_t)(n0 + s * 16 + lo) * K;  // B[k][col] = W[col][k]
      const int kb0 = kt + hi * 8;
      if (KTAIL) {
        const bool ok0 = kb0 < K;
        const bool ok1 = (kb0 + 16) < K;
        ld8g(bf[s], 0, wrow + (ok0 ? kb0 : 0), ok0);
        ld8g(bf[s], 8, wrow + (ok1 ? kb0 + 16 : 0), ok1);
      } else {
        ld8(bf[s], 0, wrow + kb0);
        ld8(bf[s], 8, wrow + kb0 + 16);
      }
    }

    // Back-to-back WMMA on independent accumulators (no D->A/B hazard).
#pragma unroll
    for (int s = 0; s < NT; ++s) {
      acc[s] = __builtin_amdgcn_wmma_f32_16x16x32_f16(
          false, af, false, bf[s], (short)0, acc[s], false, false);
    }
  }

  // Epilogue: row = tileM + hi*8 + r, col = n0 + s*16 + lo
#pragma unroll
  for (int s = 0; s < NT; ++s) {
    const int col = n0 + s * 16 + lo;
    const float bv = (bias != nullptr) ? bias[col] : 0.0f;
#pragma unroll
    for (int r = 0; r < 8; ++r) {
      const int row = tileM + hi * 8 + r;
      if (GUARD && row >= M) continue;
      float v = acc[s][r] + bv;
      const size_t idx = (size_t)row * ldc + col;
      if (EPI == 1) {
        v = siluf(v);
      } else if (EPI == 2) {
        v = softplusf(v);
      } else if (EPI == 3) {
        const float mv = (float)maskI[row];
        v = C[idx] * mv + v * mv;
      }
      C[idx] = v;
    }
  }
}

// ---------------------------------------------------------------------------
// RMSNorm per token (block = 256 threads = one token)
// ---------------------------------------------------------------------------
__global__ __launch_bounds__(256) void rmsnorm_kernel(
    const float* __restrict__ X, const float* __restrict__ w,
    const int* __restrict__ maskI, int applyMask, float* __restrict__ Y)
{
  const size_t t = blockIdx.x;
  const int    i = threadIdx.x;
  __shared__ float red[256];
  const float x = X[t * DM_ + i];
  red[i] = x * x;
  __syncthreads();
  for (int st = 128; st > 0; st >>= 1) {
    if (i < st) red[i] += red[i + st];
    __syncthreads();
  }
  const float r  = rsqrtf(red[0] / (float)DM_ + 1e-6f);
  const float mv = applyMask ? (float)maskI[t] : 1.0f;
  Y[t * DM_ + i] = x * r * w[i] * mv;
}

// ---------------------------------------------------------------------------
// Causal depthwise conv1d (width 4) + bias + SiLU; input = xz[..., :DI_]
// ---------------------------------------------------------------------------
__global__ __launch_bounds__(256) void conv_silu_kernel(
    const float* __restrict__ xz, const float* __restrict__ cw,
    const float* __restrict__ cb, float* __restrict__ xc)
{
  const size_t g = (size_t)blockIdx.x * 256 + threadIdx.x;
  if (g >= (size_t)M_ * DI_) return;
  const int    d  = (int)(g % DI_);
  const size_t bl = g / DI_;
  const int    l  = (int)(bl % L_);
  const size_t b  = bl / L_;
  float acc = cb[d];
#pragma unroll
  for (int j = 0; j < 4; ++j) {
    const int ls = l - 3 + j;
    if (ls >= 0) acc += cw[d * 4 + j] * xz[((size_t)b * L_ + ls) * (2 * DI_) + d];
  }
  xc[bl * DI_ + d] = siluf(acc);
}

// ---------------------------------------------------------------------------
// Selective scan: one lane per (b, channel d); 16 states in registers.
// Fused epilogue: y = (scan_y + u*D) * silu(z).  y may alias dt (in-place).
// ---------------------------------------------------------------------------
__global__ __launch_bounds__(64) void scan_kernel(
    const float* dt, const float* __restrict__ xz, const float* __restrict__ xc,
    const float* __restrict__ xdb, const int* __restrict__ maskI,
    const float* __restrict__ A_log, const float* __restrict__ Dp, float* y)
{
  const int b = blockIdx.x >> 3;                        // 8 d-chunks per batch
  const int d = ((blockIdx.x & 7) << 6) + threadIdx.x;  // 0..511
  float A[DS_], h[DS_];
#pragma unroll
  for (int n = 0; n < DS_; ++n) {
    A[n] = -__expf(A_log[(size_t)d * DS_ + n]);
    h[n] = 0.0f;
  }
  const float Dv = Dp[d];
  for (int l = 0; l < L_; ++l) {
    const size_t bl  = (size_t)b * L_ + l;
    const float  mv  = (float)maskI[bl];
    const float  dtv = dt[bl * DI_ + d];
    const float  u   = xc[bl * DI_ + d];
    const float* bc  = xdb + bl * XDB_;
    float yv = 0.0f;
#pragma unroll
    for (int n = 0; n < DS_; ++n) {
      const float Bt = bc[DTR_ + n];
      const float Ct = bc[DTR_ + DS_ + n];
      float hn = __expf(dtv * A[n]) * h[n] + dtv * u * Bt;
      hn *= mv;
      h[n] = hn;
      yv += hn * Ct;
    }
    const float z = xz[bl * (2 * DI_) + DI_ + d];
    y[bl * DI_ + d] = (yv + u * Dv) * siluf(z);
  }
}

// ---------------------------------------------------------------------------
// kq = k_proj^T (q_proj @ query)   (one block, 256 threads)
// ---------------------------------------------------------------------------
__global__ __launch_bounds__(256) void qkq_kernel(
    const float* __restrict__ q_proj, const float* __restrict__ query,
    const float* __restrict__ k_proj, float* __restrict__ kq)
{
  const int t = threadIdx.x;
  __shared__ float q[DM_];
  float s = 0.0f;
  for (int f = 0; f < DM_; ++f) s += q_proj[(size_t)t * DM_ + f] * query[f];
  q[t] = s;
  __syncthreads();
  float s2 = 0.0f;
  for (int e = 0; e < DM_; ++e) s2 += q[e] * k_proj[(size_t)e * DM_ + t];
  kq[t] = s2;
}

// ---------------------------------------------------------------------------
// Masked softmax pooling over L: pooled[b,:] = sum_l softmax(H.kq)[l] * H[b,l,:]
// ---------------------------------------------------------------------------
__global__ __launch_bounds__(256) void attn_pool_kernel(
    const float* __restrict__ H, const int* __restrict__ maskI,
    const float* __restrict__ kq, float* __restrict__ pooled)
{
  const int b = blockIdx.x;
  const int t = threadIdx.x;
  __shared__ float lg[L_];
  __shared__ float red[256];

  float lmax = -3.0e38f;
  for (int l = t; l < L_; l += 256) {
    const float* row = H + ((size_t)b * L_ + l) * DM_;
    float s = 0.0f;
    for (int d = 0; d < DM_; ++d) s += row[d] * kq[d];
    const bool valid = maskI[(size_t)b * L_ + l] > 0;
    s = valid ? s : -3.0e38f;
    lg[l] = s;
    lmax = fmaxf(lmax, s);
  }
  red[t] = lmax;
  __syncthreads();
  for (int st = 128; st > 0; st >>= 1) {
    if (t < st) red[t] = fmaxf(red[t], red[t + st]);
    __syncthreads();
  }
  const float mx = red[0];
  __syncthreads();

  float lsum = 0.0f;
  for (int l = t; l < L_; l += 256) {
    const bool valid = maskI[(size_t)b * L_ + l] > 0;
    const float e = valid ? __expf(lg[l] - mx) : 0.0f;
    lg[l] = e;
    lsum += e;
  }
  red[t] = lsum;
  __syncthreads();
  for (int st = 128; st > 0; st >>= 1) {
    if (t < st) red[t] += red[t + st];
    __syncthreads();
  }
  const float inv = 1.0f / red[0];

  float acc = 0.0f;
  for (int l = 0; l < L_; ++l)
    acc += lg[l] * H[((size_t)b * L_ + l) * DM_ + t];
  pooled[(size_t)b * DM_ + t] = acc * inv;
}

// ---------------------------------------------------------------------------
// attnout = out_proj( v_proj(pooled) )
// ---------------------------------------------------------------------------
__global__ __launch_bounds__(256) void attn_proj_kernel(
    const float* __restrict__ pooled, const float* __restrict__ v_proj,
    const float* __restrict__ out_proj, float* __restrict__ attnout)
{
  const int b = blockIdx.x;
  const int t = threadIdx.x;
  __shared__ float ph[DM_];
  __shared__ float pv[DM_];
  ph[t] = pooled[(size_t)b * DM_ + t];
  __syncthreads();
  float s = 0.0f;
  for (int d = 0; d < DM_; ++d) s += ph[d] * v_proj[(size_t)t * DM_ + d];
  pv[t] = s;
  __syncthreads();
  float o = 0.0f;
  for (int e = 0; e < DM_; ++e) o += pv[e] * out_proj[(size_t)t * DM_ + e];
  attnout[(size_t)b * DM_ + t] = o;
}

// ---------------------------------------------------------------------------
// last-valid-token gather + concat(last, attnout) -> cat[B, 512]
// ---------------------------------------------------------------------------
__global__ __launch_bounds__(256) void lastcat_kernel(
    const float* __restrict__ H, const int* __restrict__ maskI,
    const float* __restrict__ attnout, float* __restrict__ cat)
{
  const int b = blockIdx.x;
  const int t = threadIdx.x;
  __shared__ int red[256];
  int c = 0;
  for (int l = t; l < L_; l += 256) c += maskI[(size_t)b * L_ + l];
  red[t] = c;
  __syncthreads();
  for (int st = 128; st > 0; st >>= 1) {
    if (t < st) red[t] += red[t + st];
    __syncthreads();
  }
  const int count = red[0];
  int idx = count - 1;
  idx = idx < 0 ? 0 : (idx > L_ - 1 ? L_ - 1 : idx);
  const float lastv = (count > 0) ? H[((size_t)b * L_ + idx) * DM_ + t] : 0.0f;
  cat[(size_t)b * (2 * DM_) + t] = lastv;
  cat[(size_t)b * (2 * DM_) + DM_ + t] = attnout[(size_t)b * DM_ + t];
}

// ---------------------------------------------------------------------------
// Host launcher
// ---------------------------------------------------------------------------
extern "C" void kernel_launch(void* const* d_in, const int* in_sizes, int n_in,
                              void* d_out, int out_size, void* d_ws, size_t ws_size,
                              hipStream_t stream) {
  (void)in_sizes; (void)n_in; (void)out_size; (void)ws_size;

  // JAX tree-flatten order (dict keys alphabetical)
  const float* x          = (const float*)d_in[0];   // [B,L,16]
  const int*   mask       = (const int*)  d_in[1];   // [B,L]
  const float* attn_k     = (const float*)d_in[2];   // [256,256]
  const float* attn_outw  = (const float*)d_in[3];   // [256,256]
  const float* attn_q     = (const float*)d_in[4];   // [256,256]
  const float* attn_query = (const float*)d_in[5];   // [256]
  const float* attn_v     = (const float*)d_in[6];   // [256,256]
  const float* fusion_b   = (const float*)d_in[7];   // [256]
  const float* fusion_w   = (const float*)d_in[8];   // [256,512]
  const float* inproj_b   = (const float*)d_in[9];   // [256]
  const float* inproj_w   = (const float*)d_in[10];  // [256,16]

  float* H  = (float*)d_out;                 // [B,L,256] (= hidden state buffer)
  float* hi = H + (size_t)M_ * DM_;          // [B,256]

  float* ws      = (float*)d_ws;
  float* normed  = ws;                               // M*256
  float* xz      = normed  + (size_t)M_ * DM_;       // M*1024 (also FFN hidden)
  float* xc      = xz      + (size_t)M_ * 2 * DI_;   // M*512
  float* xdb     = xc      + (size_t)M_ * DI_;       // M*48
  float* dtb     = xdb     + (size_t)M_ * XDB_;      // M*512 (scan writes y here)
  float* kq      = dtb     + (size_t)M_ * DI_;       // 256
  float* pooled  = kq + DM_;                         // B*256
  float* attnout = pooled + B_ * DM_;                // B*256
  float* cat     = attnout + B_ * DM_;               // B*512

  const dim3 wv(32);
  const int  gm = M_ / 16;  // 1024 row tiles

  // h = x @ input_proj_w^T + input_proj_b     [M,256], K=16 (KTAIL)
  gemm16_wmma<0, 4, false, true><<<dim3(gm, DM_ / 64), wv, 0, stream>>>(
      x, FIN_, inproj_w, inproj_b, nullptr, H, DM_, M_, DM_, FIN_);

  for (int lay = 0; lay < 2; ++lay) {
    const int base = 11 + 15 * lay;
    const float* f_b1     = (const float*)d_in[base + 0];
    const float* f_b2     = (const float*)d_in[base + 1];
    const float* f_normw  = (const float*)d_in[base + 2];
    const float* f_w1     = (const float*)d_in[base + 3];   // [1024,256]
    const float* f_w2     = (const float*)d_in[base + 4];   // [256,1024]
    const float* m_Alog   = (const float*)d_in[base + 5];   // [512,16]
    const float* m_D      = (const float*)d_in[base + 6];   // [512]
    const float* m_convb  = (const float*)d_in[base + 7];   // [512]
    const float* m_convw  = (const float*)d_in[base + 8];   // [512,1,4]
    const float* m_dtb    = (const float*)d_in[base + 9];   // [512]
    const float* m_dtw    = (const float*)d_in[base + 10];  // [512,16]
    const float* m_inprj  = (const float*)d_in[base + 11];  // [1024,256]
    const float* m_normw  = (const float*)d_in[base + 12];  // [256]
    const float* m_outprj = (const float*)d_in[base + 13];  // [256,512]
    const float* m_xprj   = (const float*)d_in[base + 14];  // [48,512]

    // --- Mamba block ---
    rmsnorm_kernel<<<M_, 256, 0, stream>>>(H, m_normw, mask, 1, normed);
    gemm16_wmma<0, 4, false, false><<<dim3(gm, (2 * DI_) / 64), wv, 0, stream>>>(
        normed, DM_, m_inprj, nullptr, nullptr, xz, 2 * DI_, M_, 2 * DI_, DM_);
    conv_silu_kernel<<<cdiv(M_ * DI_, 256), 256, 0, stream>>>(xz, m_convw, m_convb, xc);
    gemm16_wmma<0, 3, false, false><<<dim3(gm, 1), wv, 0, stream>>>(
        xc, DI_, m_xprj, nullptr, nullptr, xdb, XDB_, M_, XDB_, DI_);
    gemm16_wmma<2, 4, false, true><<<dim3(gm, DI_ / 64), wv, 0, stream>>>(
        xdb, XDB_, m_dtw, m_dtb, nullptr, dtb, DI_, M_, DI_, DTR_);
    scan_kernel<<<B_ * 8, 64, 0, stream>>>(dtb, xz, xc, xdb, mask, m_Alog, m_D, dtb);
    gemm16_wmma<3, 4, false, false><<<dim3(gm, DM_ / 64), wv, 0, stream>>>(
        dtb, DI_, m_outprj, nullptr, mask, H, DM_, M_, DM_, DI_);

    // --- FFN block ---
    rmsnorm_kernel<<<M_, 256, 0, stream>>>(H, f_normw, mask, 0, normed);
    gemm16_wmma<1, 4, false, false><<<dim3(gm, (4 * DM_) / 64), wv, 0, stream>>>(
        normed, DM_, f_w1, f_b1, nullptr, xz, 4 * DM_, M_, 4 * DM_, DM_);
    gemm16_wmma<3, 4, false, false><<<dim3(gm, DM_ / 64), wv, 0, stream>>>(
        xz, 4 * DM_, f_w2, f_b2, mask, H, DM_, M_, DM_, 4 * DM_);
  }

  // --- pooling head ---
  qkq_kernel<<<1, 256, 0, stream>>>(attn_q, attn_query, attn_k, kq);
  attn_pool_kernel<<<B_, 256, 0, stream>>>(H, mask, kq, pooled);
  attn_proj_kernel<<<B_, 256, 0, stream>>>(pooled, attn_v, attn_outw, attnout);
  lastcat_kernel<<<B_, 256, 0, stream>>>(H, mask, attnout, cat);

  // h_i = cat @ fusion_w^T + fusion_b        [8,256], K=512 (row-GUARD, M=8)
  gemm16_wmma<0, 4, true, false><<<dim3(1, DM_ / 64), wv, 0, stream>>>(
      cat, 2 * DM_, fusion_w, fusion_b, nullptr, hi, DM_, B_, DM_, 2 * DM_);
}
